// ClassConditionalLM_85452669321722
// MI455X (gfx1250) — matrix operand order repaired and checked by reading, not compiled, not run on hardware
//
#include <hip/hip_runtime.h>
#include <hip/hip_fp16.h>

typedef __attribute__((ext_vector_type(16))) _Float16 v16h;
typedef __attribute__((ext_vector_type(8)))  _Float16 v8h;
typedef __attribute__((ext_vector_type(4)))  _Float16 v4h;
typedef __attribute__((ext_vector_type(8)))  float    v8f;

#define N_ROWS        131072
#define LFS           128
#define CLS           64
#define KCOLS         80      // 64 classes + prop col (64) + ones col (65) + pad, 5 tiles of 16
#define SW            68      // padded scatter/corr row width: 0..63 classes, 66 = trash slot
#define WAVES_PER_WG  8
#define ROWS_PER_WAVE 16
#define ROWS_PER_WG   (WAVES_PER_WG * ROWS_PER_WAVE)  // 128
#define THREADS       (WAVES_PER_WG * 32)             // 256

// Precomputed tables: built once by the prep kernel (or per-WG fallback).
struct Tables {
    _Float16 Wt[KCOLS][LFS];   // 20480 B  B: [k][l]; k<64: z_acc+acc, 64: prop, 65: 1.0
    float    corr[LFS][SW];    // 34816 B  corr[l][k] = 2*acc + log(K-1); k>=64: 0
    float    prior[CLS];       //   256 B  log class prior
    float    zp;               // sum of logaddexp(prop, 0)
    float    pad[3];
};
#define TABLE_U4 (sizeof(Tables) / 16)   // 3473 uint4s

struct SMem {
    Tables   t;
    _Float16 mbuf[WAVES_PER_WG][ROWS_PER_WAVE][LFS];   // 32 KB  f16 mask tiles (wave-private)
    float    sbuf[WAVES_PER_WG][ROWS_PER_WAVE][SW];    // 34 KB  cll tiles (slot 66 = trash)
    float    srow[WAVES_PER_WG][ROWS_PER_WAVE];
    float    prop[LFS];                                // fallback path only
};

// ---------------- one-shot table builder ----------------
__global__ void cc_prep(const float* __restrict__ accuracy,
                        const float* __restrict__ propensity,
                        const float* __restrict__ class_balance,
                        Tables* __restrict__ tab) {
    __shared__ float sprop[LFS];
    __shared__ float sprior[CLS];
    const int tid = threadIdx.x;
    const float logKm1 = __logf((float)(CLS - 1));
    if (tid < LFS) sprop[tid]  = propensity[tid];
    if (tid < CLS) sprior[tid] = class_balance[tid];
    __syncthreads();
    if (tid == 0) {
        float m = sprior[0];
        for (int k = 1; k < CLS; ++k) m = fmaxf(m, sprior[k]);
        float a = 0.f;
        for (int k = 0; k < CLS; ++k) a += __expf(sprior[k] - m);
        const float lse = m + __logf(a);
        for (int k = 0; k < CLS; ++k) tab->prior[k] = sprior[k] - lse;
        float zp = 0.f;
        for (int l = 0; l < LFS; ++l) {
            const float p = sprop[l];
            zp += fmaxf(p, 0.f) + __logf(1.f + __expf(-fabsf(p)));
        }
        tab->zp = zp;
    }
    for (int i = tid; i < KCOLS * LFS; i += THREADS) {
        const int k = i >> 7;
        const int l = i & (LFS - 1);
        _Float16 w = (_Float16)0.0f;
        if (k < CLS) {
            const float a  = accuracy[l * CLS + k];
            const float za = fabsf(a) + __logf(1.f + __expf(-2.f * fabsf(a)));
            w = (_Float16)(za + a);
        } else if (k == CLS)     w = (_Float16)sprop[l];
        else if (k == CLS + 1)   w = (_Float16)1.0f;
        tab->Wt[k][l] = w;
    }
    for (int i = tid; i < LFS * SW; i += THREADS) {
        const int l = i / SW;
        const int k = i - l * SW;
        tab->corr[l][k] = (k < CLS) ? (2.f * accuracy[l * CLS + k] + logKm1) : 0.f;
    }
}

// ---------------- main kernel (single instantiation; tab==nullptr -> build tables locally) ----------------
__global__ __launch_bounds__(THREADS, 1)
void ClassConditionalLM_kernel(const int*    __restrict__ votes,
                               const float*  __restrict__ accuracy,
                               const float*  __restrict__ propensity,
                               const float*  __restrict__ class_balance,
                               const Tables* __restrict__ tab,
                               float*        __restrict__ out) {
    extern __shared__ char smem_raw[];
    SMem& s = *reinterpret_cast<SMem*>(smem_raw);

    const int tid  = threadIdx.x;
    const int wave = tid >> 5;
    const int lane = tid & 31;
    const int h    = lane >> 4;
    const int r    = lane & 15;
    const float logKm1 = __logf((float)(CLS - 1));

    const long long rowBase = (long long)blockIdx.x * ROWS_PER_WG + (long long)wave * ROWS_PER_WAVE;
    const int* vtile = votes + rowBase * LFS;

    #pragma unroll
    for (int i = 0; i < ROWS_PER_WAVE; ++i)
        __builtin_prefetch(vtile + i * LFS + lane * 4, 0, 3);

    // ---------------- tables into LDS ----------------
    if (tab != nullptr) {               // uniform branch (SALU)
        const uint4* src = (const uint4*)tab;
        uint4*       dst = (uint4*)&s.t;
        for (int i = tid; i < (int)TABLE_U4; i += THREADS) dst[i] = src[i];
    } else {
        if (tid < LFS) s.prop[tid]    = propensity[tid];
        if (tid < CLS) s.t.prior[tid] = class_balance[tid];
        __syncthreads();
        if (tid == 0) {
            float m = s.t.prior[0];
            for (int k = 1; k < CLS; ++k) m = fmaxf(m, s.t.prior[k]);
            float a = 0.f;
            for (int k = 0; k < CLS; ++k) a += __expf(s.t.prior[k] - m);
            const float lse = m + __logf(a);
            for (int k = 0; k < CLS; ++k) s.t.prior[k] -= lse;
            float zp = 0.f;
            for (int l = 0; l < LFS; ++l) {
                const float p = s.prop[l];
                zp += fmaxf(p, 0.f) + __logf(1.f + __expf(-fabsf(p)));
            }
            s.t.zp = zp;
        }
        for (int i = tid; i < KCOLS * LFS; i += THREADS) {
            const int k = i >> 7;
            const int l = i & (LFS - 1);
            _Float16 w = (_Float16)0.0f;
            if (k < CLS) {
                const float a  = accuracy[l * CLS + k];
                const float za = fabsf(a) + __logf(1.f + __expf(-2.f * fabsf(a)));
                w = (_Float16)(za + a);
            } else if (k == CLS)     w = (_Float16)s.prop[l];
            else if (k == CLS + 1)   w = (_Float16)1.0f;
            s.t.Wt[k][l] = w;
        }
        for (int i = tid; i < LFS * SW; i += THREADS) {
            const int l = i / SW;
            const int k = i - l * SW;
            s.t.corr[l][k] = (k < CLS) ? (2.f * accuracy[l * CLS + k] + logKm1) : 0.f;
        }
    }
    __syncthreads();

    // ---------------- init scatter tile rows with class prior (b128 stores) ----------------
    {
        float4 pv = make_float4(0.f, 0.f, 0.f, 0.f);
        if (lane < 16) pv = *(const float4*)&s.t.prior[lane * 4];
        if (lane < 17) {
            #pragma unroll
            for (int row = 0; row < ROWS_PER_WAVE; ++row)
                *(float4*)&s.sbuf[wave][row][lane * 4] = pv;   // lane16 zeros slots 64..67
        }
    }
    asm volatile("s_wait_dscnt 0" ::: "memory");

    // ---------------- fused staging: f16 mask store + branch-free scatter from registers ----------------
    {
        const float* corrRow = &s.t.corr[lane * 4][0];         // this lane covers l = lane*4 .. lane*4+3
        float*       sbufw   = &s.sbuf[wave][0][0];
        #pragma unroll
        for (int i = 0; i < ROWS_PER_WAVE; ++i) {
            const int4 v4 = ((const int4*)(vtile + i * LFS))[lane];
            v4h m;
            m[0] = v4.x ? (_Float16)1.0f : (_Float16)0.0f;
            m[1] = v4.y ? (_Float16)1.0f : (_Float16)0.0f;
            m[2] = v4.z ? (_Float16)1.0f : (_Float16)0.0f;
            m[3] = v4.w ? (_Float16)1.0f : (_Float16)0.0f;
            *(v4h*)&s.mbuf[wave][i][lane * 4] = m;
            const int vv[4] = { v4.x, v4.y, v4.z, v4.w };
            #pragma unroll
            for (int e = 0; e < 4; ++e) {
                unsigned int iu = (unsigned int)vv[e] - 1u;    // v==0 underflows -> trash slot
                iu = iu < 66u ? iu : 66u;                      // v_min_u32
                atomicAdd(&sbufw[i * SW + iu], corrRow[e * SW + iu]);   // ds_add_f32
            }
        }
    }
    asm volatile("s_wait_dscnt 0" ::: "memory");

    // ---------------- A fragments: pure ds_load_b128 from mask tile ----------------
    v16h afrag[4];
    {
        const _Float16* mrow = &s.mbuf[wave][r][0];
        #pragma unroll
        for (int c = 0; c < 4; ++c) {
            const v8h lo = *(const v8h*)(mrow + c * 32 + h * 8);
            const v8h hi = *(const v8h*)(mrow + c * 32 + 16 + h * 8);
            afrag[c] = __builtin_shufflevector(lo, hi,
                            0, 1, 2, 3, 4, 5, 6, 7, 8, 9, 10, 11, 12, 13, 14, 15);
        }
    }

    // ---------------- WMMA: 5 class tiles (incl. prop/ones columns) x 4 K-steps ----------------
    v8f acc[5];
    #pragma unroll
    for (int kt = 0; kt < 5; ++kt) {
        v8f c8 = {};
        #pragma unroll
        for (int c = 0; c < 4; ++c) {
            const _Float16* bp = &s.t.Wt[kt * 16 + r][c * 32];
            const v8h blo = *(const v8h*)(bp + h * 8);
            const v8h bhi = *(const v8h*)(bp + 16 + h * 8);
            const v16h bf = __builtin_shufflevector(blo, bhi,
                                0, 1, 2, 3, 4, 5, 6, 7, 8, 9, 10, 11, 12, 13, 14, 15);
            c8 = __builtin_amdgcn_wmma_f32_16x16x32_f16(false, afrag[c], false, bf,
                                                        (short)0, c8, false, false);
        }
        acc[kt] = c8;
    }

    // ---------------- per-row scalar term from tile 4 (pdot in r=0, cnt in r=1) ----------------
    {
        const float zp = s.t.zp;
        #pragma unroll
        for (int j = 0; j < 8; ++j) {
            const float other = __shfl_xor(acc[4][j], 1, 32);
            const float srowv = -zp + acc[4][j] - other * logKm1;
            if (r == 0) s.srow[wave][j + 8 * h] = srowv;
        }
    }
    asm volatile("s_wait_dscnt 0" ::: "memory");

    // ---------------- fold (s_row - mm) into scatter tile ----------------
    {
        const float4 sr0 = *(const float4*)&s.srow[wave][8 * h];
        const float4 sr1 = *(const float4*)&s.srow[wave][8 * h + 4];
        const float srv[8] = { sr0.x, sr0.y, sr0.z, sr0.w, sr1.x, sr1.y, sr1.z, sr1.w };
        #pragma unroll
        for (int kt = 0; kt < 4; ++kt)
            #pragma unroll
            for (int j = 0; j < 8; ++j)
                atomicAdd(&s.sbuf[wave][j + 8 * h][kt * 16 + r], srv[j] - acc[kt][j]);
    }
    asm volatile("s_wait_dscnt 0" ::: "memory");

    // ---------------- per-row logsumexp over 64 classes ----------------
    {
        const float4* rp = (const float4*)&s.sbuf[wave][r][h * 32];
        float4 x[8];
        #pragma unroll
        for (int j = 0; j < 8; ++j) x[j] = rp[j];
        float m = -3.0e38f;
        #pragma unroll
        for (int j = 0; j < 8; ++j)
            m = fmaxf(m, fmaxf(fmaxf(x[j].x, x[j].y), fmaxf(x[j].z, x[j].w)));
        float ssum = 0.f;
        #pragma unroll
        for (int j = 0; j < 8; ++j)
            ssum += __expf(x[j].x - m) + __expf(x[j].y - m) +
                    __expf(x[j].z - m) + __expf(x[j].w - m);
        const float m2 = __shfl_xor(m,    16, 32);
        const float s2 = __shfl_xor(ssum, 16, 32);
        const float M  = fmaxf(m, m2);
        const float S  = ssum * __expf(m - M) + s2 * __expf(m2 - M);
        if (h == 0) out[rowBase + r] = M + __logf(S);
    }
}

extern "C" void kernel_launch(void* const* d_in, const int* in_sizes, int n_in,
                              void* d_out, int out_size, void* d_ws, size_t ws_size,
                              hipStream_t stream) {
    const int*   votes         = (const int*)d_in[0];
    const float* accuracy      = (const float*)d_in[1];
    const float* propensity    = (const float*)d_in[2];
    const float* class_balance = (const float*)d_in[3];
    float*       out           = (float*)d_out;

    const dim3 grid(N_ROWS / ROWS_PER_WG);   // 1024 workgroups
    const dim3 block(THREADS);               // 8 waves (wave32)

    Tables* tab = (ws_size >= sizeof(Tables)) ? (Tables*)d_ws : nullptr;
    if (tab)
        cc_prep<<<dim3(1), block, 0, stream>>>(accuracy, propensity, class_balance, tab);
    ClassConditionalLM_kernel<<<grid, block, sizeof(SMem), stream>>>(
        votes, accuracy, propensity, class_balance, tab, out);
}
